// AttentionBlock_14302241096180
// MI455X (gfx1250) — matrix operand order repaired and checked by reading, not compiled, not run on hardware
//
#include <hip/hip_runtime.h>

// ---------------------------------------------------------------------------
// Attention block for MI455X (gfx1250, wave32, WMMA).
//   Q = x Wq ; K = x Wk ; V = x Wv    (bf16 operands, f32 accum via v_wmma)
//   attn = softmax(QK^T / sqrt(D)) V  (flash-style online softmax)
//   out  = concat([x, attn], -1)
//
//   * batched fragment loads + sched_barrier so the backend keeps all
//     fragments in distinct registers -> clause loads, partial waits
//   * V tiles prefetched into LDS with GLOBAL_LOAD_ASYNC_TO_LDS_B128
//     (ASYNCcnt path), overlapped with the score+softmax phase
// Workspace: xb 16MB + Qb 16MB + Kb 16MB + Vt 16MB + Wt 6MB = 70 MB.
// ---------------------------------------------------------------------------

#define BATCH 4
#define SEQ   2048
#define DIM   1024

typedef __attribute__((ext_vector_type(16))) __bf16  v16bf;
typedef __attribute__((ext_vector_type(8)))  float   v8f;
typedef __attribute__((ext_vector_type(4)))  float   f32x4;
typedef __attribute__((ext_vector_type(4)))  unsigned int u32x4;
typedef int i32x4v __attribute__((__vector_size__(4 * sizeof(int))));

#if defined(__HIP_DEVICE_COMPILE__) && __has_builtin(__builtin_amdgcn_global_load_async_to_lds_b128)
#define USE_ASYNC_LDS 1
#else
#define USE_ASYNC_LDS 0
#endif

union BF16x16 {           // one 16x16x32 bf16 WMMA operand fragment (8 VGPRs)
    v16bf          v;
    unsigned short u[16];
    u32x4          q[2];
};

__device__ inline unsigned short f2bf(float f) {
    union { float f; unsigned int u; } x; x.f = f;
    unsigned int r = x.u + 0x7FFFu + ((x.u >> 16) & 1u);   // round-nearest-even
    return (unsigned short)(r >> 16);
}

__device__ inline v8f wmma_bf16(const v16bf a, const v16bf b, const v8f c) {
    return __builtin_amdgcn_wmma_f32_16x16x32_bf16(
        false, a, false, b, (short)0, c, false, false);
}

// Scheduling fence: keep the load batch before it from sinking into the
// compute batch after it (forces distinct registers + clause + partial waits).
__device__ inline void sched_fence() {
#if __has_builtin(__builtin_amdgcn_sched_barrier)
    __builtin_amdgcn_sched_barrier(0);
#endif
}

#if USE_ASYNC_LDS
__device__ inline void async_ld16(const unsigned short* src, unsigned short* lds_dst) {
    // 16B per lane: LDS[lds_dst] = MEM[src]; tracked by ASYNCcnt
    __builtin_amdgcn_global_load_async_to_lds_b128(
        (__attribute__((address_space(1))) i32x4v*)const_cast<unsigned short*>(src),
        (__attribute__((address_space(3))) i32x4v*)lds_dst,
        0, 0);
}
__device__ inline void wait_async0() {
#if __has_builtin(__builtin_amdgcn_s_wait_asynccnt)
    __builtin_amdgcn_s_wait_asynccnt(0);
#else
    asm volatile("s_wait_asynccnt 0x0" ::: "memory");
#endif
}
#endif

// ---------------------------------------------------------------------------
// Prep 1: x fp32 -> bf16 (contiguous, vectorized: 8 elems / thread)
// ---------------------------------------------------------------------------
__global__ __launch_bounds__(256) void prep_x_kernel(
    const float* __restrict__ x, unsigned short* __restrict__ xb)
{
    const size_t total = (size_t)BATCH * SEQ * DIM / 8;
    const size_t i = (size_t)blockIdx.x * blockDim.x + threadIdx.x;
    if (i >= total) return;
    const float* p = x + i * 8;
    f32x4 a = *(const f32x4*)(p);
    f32x4 b = *(const f32x4*)(p + 4);
    union { unsigned short u[8]; u32x4 q; } o;
    o.u[0] = f2bf(a.x); o.u[1] = f2bf(a.y); o.u[2] = f2bf(a.z); o.u[3] = f2bf(a.w);
    o.u[4] = f2bf(b.x); o.u[5] = f2bf(b.y); o.u[6] = f2bf(b.z); o.u[7] = f2bf(b.w);
    *(u32x4*)(xb + i * 8) = o.q;
}

// ---------------------------------------------------------------------------
// Prep 2: W[k][n] fp32 -> Wt[n][k] bf16, LDS-tiled 32x32 transpose.
// ---------------------------------------------------------------------------
__global__ __launch_bounds__(256) void prep_w_kernel(
    const float* __restrict__ Wq, const float* __restrict__ Wk,
    const float* __restrict__ Wv, unsigned short* __restrict__ Wqt,
    unsigned short* __restrict__ Wkt, unsigned short* __restrict__ Wvt)
{
    __shared__ float tile[32][33];
    const float* W = (blockIdx.y == 0) ? Wq : (blockIdx.y == 1) ? Wk : Wv;
    unsigned short* Wt = (blockIdx.y == 0) ? Wqt : (blockIdx.y == 1) ? Wkt : Wvt;

    const int NT = DIM / 32;
    const int kt = (blockIdx.x / NT) * 32;
    const int nt = (blockIdx.x % NT) * 32;
    const int tx = threadIdx.x & 31;
    const int ty = threadIdx.x >> 5;
#pragma unroll
    for (int i = 0; i < 4; ++i)
        tile[ty + i * 8][tx] = W[(size_t)(kt + ty + i * 8) * DIM + nt + tx];
    __syncthreads();
#pragma unroll
    for (int i = 0; i < 4; ++i)
        Wt[(size_t)(nt + ty + i * 8) * DIM + kt + tx] = f2bf(tile[tx][ty + i * 8]);
}

// ---------------------------------------------------------------------------
// Kernel 1: QKV projection. Each wave: 16x64 tile of Q, K and V.
// Per K-step: batch-load 12 B fragments + 1 A fragment, fence, 12 WMMAs.
// ---------------------------------------------------------------------------
__global__ __launch_bounds__(256, 2) void qkv_proj_kernel(
    const unsigned short* __restrict__ xb,
    const unsigned short* __restrict__ Wqt, const unsigned short* __restrict__ Wkt,
    const unsigned short* __restrict__ Wvt,
    unsigned short* __restrict__ Qb, unsigned short* __restrict__ Kb,
    unsigned short* __restrict__ Vt)
{
    const int wave = threadIdx.x >> 5;
    const int lane = threadIdx.x & 31;
    const int h = lane >> 4, m = lane & 15, n = m;

    const int COLG = DIM / 64;
    const int row0 = (blockIdx.x / COLG) * 128 + wave * 16;
    const int colb = (blockIdx.x % COLG) * 64;

    v8f accQ[4], accK[4], accV[4];
#pragma unroll
    for (int t = 0; t < 4; ++t) { accQ[t] = (v8f){}; accK[t] = (v8f){}; accV[t] = (v8f){}; }

    for (int kk = 0; kk < DIM; kk += 32) {
        BF16x16 a, bq[4], bk[4], bv[4];
        // ---- issue ALL loads for this step ----
        const unsigned short* xr = xb + (size_t)(row0 + m) * DIM + kk + h * 16;
        a.q[0] = *(const u32x4*)(xr);
        a.q[1] = *(const u32x4*)(xr + 8);
#pragma unroll
        for (int t = 0; t < 4; ++t) {
            const size_t wo = (size_t)(colb + t * 16 + n) * DIM + kk + h * 16;
            bq[t].q[0] = *(const u32x4*)(Wqt + wo); bq[t].q[1] = *(const u32x4*)(Wqt + wo + 8);
            bk[t].q[0] = *(const u32x4*)(Wkt + wo); bk[t].q[1] = *(const u32x4*)(Wkt + wo + 8);
            bv[t].q[0] = *(const u32x4*)(Wvt + wo); bv[t].q[1] = *(const u32x4*)(Wvt + wo + 8);
        }
        sched_fence();                                  // loads must not sink
        // ---- 12 WMMAs sharing the A fragment ----
#pragma unroll
        for (int t = 0; t < 4; ++t) {
            accQ[t] = wmma_bf16(a.v, bq[t].v, accQ[t]);
            accK[t] = wmma_bf16(a.v, bk[t].v, accK[t]);
            accV[t] = wmma_bf16(a.v, bv[t].v, accV[t]);
        }
    }

    // C layout: vgpr r -> row h*8+r, column n. V stored transposed [b][d][s].
#pragma unroll
    for (int t = 0; t < 4; ++t) {
#pragma unroll
        for (int r = 0; r < 8; ++r) {
            const int row = row0 + h * 8 + r;
            const int col = colb + t * 16 + n;
            const size_t off = (size_t)row * DIM + col;
            Qb[off] = f2bf(accQ[t][r]);
            Kb[off] = f2bf(accK[t][r]);
            const int b = row / SEQ, s = row % SEQ;
            Vt[((size_t)b * DIM + col) * SEQ + s] = f2bf(accV[t][r]);
        }
    }
}

// ---------------------------------------------------------------------------
// Kernel 2: flash attention. Block = 8 waves handles (batch, 16-query tile).
// Wave w owns head-dim slice [w*128,(w+1)*128). V slice async-prefetched into
// LDS at step top (ASYNCcnt), consumed after softmax (latency fully hidden).
// ---------------------------------------------------------------------------
__global__ __launch_bounds__(256, 2) void attn_kernel(
    const unsigned short* __restrict__ Qb, const unsigned short* __restrict__ Kb,
    const unsigned short* __restrict__ Vt, float* __restrict__ out)
{
    __shared__ float          S_tile[16][32];
    __shared__ unsigned short P_tile[16][32];
    __shared__ float          m_sh[16], l_sh[16], f_sh[16];
#if USE_ASYNC_LDS
    extern __shared__ unsigned short Vbuf[];   // [wave][t][lane][16] = 64KB
#endif

    const int wave = threadIdx.x >> 5;
    const int lane = threadIdx.x & 31;
    const int h = lane >> 4, m = lane & 15, n = m;

    const int    b     = blockIdx.x / (SEQ / 16);
    const int    q0    = (blockIdx.x % (SEQ / 16)) * 16;
    const size_t base  = (size_t)b * SEQ * DIM;
    const size_t vbase = (size_t)b * DIM * SEQ;
    const int    d0    = wave * 128;
    const float  scale = 0.03125f;                     // 1/sqrt(1024)

    // Resident Q A-fragments for this wave's 128-dim slice (4 x K=32)
    BF16x16 qf[4];
#pragma unroll
    for (int c = 0; c < 4; ++c) {
        const unsigned short* p = Qb + base + (size_t)(q0 + m) * DIM + d0 + c * 32 + h * 16;
        qf[c].q[0] = *(const u32x4*)(p);
        qf[c].q[1] = *(const u32x4*)(p + 8);
    }

    v8f acc[8];
#pragma unroll
    for (int t = 0; t < 8; ++t) acc[t] = (v8f){};
    if (threadIdx.x < 16) { m_sh[threadIdx.x] = -1e30f; l_sh[threadIdx.x] = 0.0f; }

    for (int key0 = 0; key0 < SEQ; key0 += 32) {
#if USE_ASYNC_LDS
        // ---- async prefetch of this step's V slice (overlaps score phase) ----
#pragma unroll
        for (int t = 0; t < 8; ++t) {
            const unsigned short* vp =
                Vt + vbase + (size_t)(d0 + t * 16 + n) * SEQ + key0 + h * 16;
            unsigned short* ld = Vbuf + (((size_t)wave * 8 + t) * 32 + lane) * 16;
            async_ld16(vp,     ld);
            async_ld16(vp + 8, ld + 8);
        }
#endif
        __syncthreads();                               // prior P/f reads done
        for (int i = threadIdx.x; i < 16 * 32; i += 256) ((float*)S_tile)[i] = 0.0f;
        __syncthreads();

        // ---- partial scores: batch-load 8 K fragments, fence, 8 WMMAs ----
        BF16x16 kf[8];
#pragma unroll
        for (int c = 0; c < 4; ++c) {
            const unsigned short* kp0 = Kb + base + (size_t)(key0 + n) * DIM + d0 + c * 32 + h * 16;
            const unsigned short* kp1 = kp0 + (size_t)16 * DIM;
            kf[c].q[0]     = *(const u32x4*)(kp0); kf[c].q[1]     = *(const u32x4*)(kp0 + 8);
            kf[c + 4].q[0] = *(const u32x4*)(kp1); kf[c + 4].q[1] = *(const u32x4*)(kp1 + 8);
        }
        sched_fence();
        v8f s0 = {}, s1 = {};
#pragma unroll
        for (int c = 0; c < 4; ++c) {
            s0 = wmma_bf16(qf[c].v, kf[c].v,     s0);
            s1 = wmma_bf16(qf[c].v, kf[c + 4].v, s1);
        }
#pragma unroll
        for (int r = 0; r < 8; ++r) {                  // cross-wave reduction
            atomicAdd(&S_tile[h * 8 + r][n],      s0[r]);
            atomicAdd(&S_tile[h * 8 + r][16 + n], s1[r]);
        }
        __syncthreads();

        // ---- online softmax: one thread per query row ----
        if (threadIdx.x < 16) {
            const int row = threadIdx.x;
            const float mprev = m_sh[row];
            float sv[32], mx = mprev;
#pragma unroll
            for (int c2 = 0; c2 < 32; ++c2) {
                sv[c2] = S_tile[row][c2] * scale;
                mx = fmaxf(mx, sv[c2]);
            }
            const float f = __expf(mprev - mx);
            float sum = 0.0f;
#pragma unroll
            for (int c2 = 0; c2 < 32; ++c2) {
                const float p = __expf(sv[c2] - mx);
                sum += p;
                P_tile[row][c2] = f2bf(p);
            }
            l_sh[row] = l_sh[row] * f + sum;
            m_sh[row] = mx;
            f_sh[row] = f;
        }
        __syncthreads();                               // P_tile / f_sh ready

        // ---- rescale accumulators ----
        float fr[8];
#pragma unroll
        for (int r = 0; r < 8; ++r) fr[r] = f_sh[h * 8 + r];
#pragma unroll
        for (int t = 0; t < 8; ++t)
#pragma unroll
            for (int r = 0; r < 8; ++r) acc[t][r] *= fr[r];

        BF16x16 pf;
        {
            const unsigned short* pp = &P_tile[m][h * 16];
            pf.q[0] = *(const u32x4*)(pp);
            pf.q[1] = *(const u32x4*)(pp + 8);
        }

        // ---- P (16x32) . V : batch V fragments, fence, 8 WMMAs ----
        BF16x16 vf[8];
#if USE_ASYNC_LDS
        wait_async0();                                 // this wave's DMA done
#pragma unroll
        for (int t = 0; t < 8; ++t) {
            const unsigned short* ld = Vbuf + (((size_t)wave * 8 + t) * 32 + lane) * 16;
            vf[t].q[0] = *(const u32x4*)(ld);
            vf[t].q[1] = *(const u32x4*)(ld + 8);
        }
#else
#pragma unroll
        for (int t = 0; t < 8; ++t) {
            const unsigned short* vp =
                Vt + vbase + (size_t)(d0 + t * 16 + n) * SEQ + key0 + h * 16;
            vf[t].q[0] = *(const u32x4*)(vp);
            vf[t].q[1] = *(const u32x4*)(vp + 8);
        }
#endif
        sched_fence();
#pragma unroll
        for (int t = 0; t < 8; ++t)
            acc[t] = wmma_bf16(pf.v, vf[t].v, acc[t]);
    }
    __syncthreads();                                   // final l_sh stable

    // ---- epilogue: attn = acc / l ; write to out[..., D:2D] ----
    float lr[8];
#pragma unroll
    for (int r = 0; r < 8; ++r) lr[r] = 1.0f / l_sh[h * 8 + r];
#pragma unroll
    for (int t = 0; t < 8; ++t)
#pragma unroll
        for (int r = 0; r < 8; ++r) {
            const int row = q0 + h * 8 + r;
            const int col = d0 + t * 16 + n;
            out[((size_t)b * SEQ + row) * (2 * DIM) + DIM + col] = acc[t][r] * lr[r];
        }
}

// ---------------------------------------------------------------------------
// Kernel 3: out[..., 0:D] = x  (vectorized strided copy)
// ---------------------------------------------------------------------------
__global__ __launch_bounds__(256) void copy_x_kernel(
    const float* __restrict__ x, float* __restrict__ out)
{
    const size_t total = (size_t)BATCH * SEQ * DIM / 4;
    const size_t i = (size_t)blockIdx.x * blockDim.x + threadIdx.x;
    if (i >= total) return;
    const size_t e   = i * 4;
    const size_t row = e / DIM;
    const size_t c   = e % DIM;
    *(f32x4*)(out + row * (2 * DIM) + c) = *(const f32x4*)(x + e);
}

// ---------------------------------------------------------------------------
extern "C" void kernel_launch(void* const* d_in, const int* in_sizes, int n_in,
                              void* d_out, int out_size, void* d_ws, size_t ws_size,
                              hipStream_t stream) {
    const float* x  = (const float*)d_in[0];
    const float* Wq = (const float*)d_in[1];
    const float* Wk = (const float*)d_in[2];
    const float* Wv = (const float*)d_in[3];
    float* out = (float*)d_out;

    const size_t E  = (size_t)BATCH * SEQ * DIM;       // 8M elems
    const size_t WE = (size_t)DIM * DIM;               // 1M elems
    unsigned short* xb  = (unsigned short*)d_ws;       // 70 MB total workspace
    unsigned short* Qb  = xb + E;
    unsigned short* Kb  = Qb + E;
    unsigned short* Vt  = Kb + E;
    unsigned short* Wqt = Vt + E;
    unsigned short* Wkt = Wqt + WE;
    unsigned short* Wvt = Wkt + WE;

    prep_x_kernel<<<(int)(E / 8 / 256), 256, 0, stream>>>(x, xb);
    {
        dim3 g((DIM / 32) * (DIM / 32), 3);
        prep_w_kernel<<<g, 256, 0, stream>>>(Wq, Wk, Wv, Wqt, Wkt, Wvt);
    }
    {
        const int blocks = ((BATCH * SEQ) / 128) * (DIM / 64);
        qkv_proj_kernel<<<blocks, 256, 0, stream>>>(xb, Wqt, Wkt, Wvt, Qb, Kb, Vt);
    }
    // 64KB dynamic LDS for the async V staging buffer
    attn_kernel<<<BATCH * (SEQ / 16), 256, 8 * 8 * 32 * 16 * sizeof(unsigned short), stream>>>(
        Qb, Kb, Vt, out);
    {
        const size_t total = (size_t)BATCH * SEQ * DIM / 4;
        copy_x_kernel<<<(int)((total + 255) / 256), 256, 0, stream>>>(x, out);
    }
}
